// Attention_51642686767948
// MI455X (gfx1250) — compile-verified, matrix operand order checked by reading, not run on hardware
//
#include <hip/hip_runtime.h>

// ---------------- Problem constants ----------------
#define BB 8
#define CC 16
#define LL 256
#define DD 512
#define MM (BB * CC * LL)          // 32768 rows
#define NQKV (3 * DD)              // 1536
#define SCALE_F 0.125f             // 64^-0.5

// ---------------- Types ----------------
typedef __attribute__((ext_vector_type(16))) __bf16   v16bf;
typedef __attribute__((ext_vector_type(8)))  float    v8f;
typedef __attribute__((ext_vector_type(4)))  unsigned int uint4v;

union FragU { v16bf v; unsigned int u[8]; unsigned short us[16]; };

__device__ __forceinline__ unsigned short f2bf(float x) {
    unsigned int u = __float_as_uint(x);
    u += 0x7FFFu + ((u >> 16) & 1u);          // round-to-nearest-even
    return (unsigned short)(u >> 16);
}

// ---- WMMA wrapper: D = A(16x32 bf16) * B(32x16 bf16) + C(f32) ----
__device__ __forceinline__ v8f wmma_bf16(v16bf a, v16bf b, v8f c) {
    return __builtin_amdgcn_wmma_f32_16x16x32_bf16(
        false, a, false, b, (short)0, c, false, false);
}

// ---- Fragment loader ----
// 16-bit A-matrix 16x32 layout (ISA 7.12.2): lanes 0-15 hold K {0..7,16..23},
// lanes 16-31 hold K {8..15,24..31}; M (or N for B-as-BT) = lane & 15.
// Row-major bf16 source; two b128 loads per lane, no VALU.
__device__ __forceinline__ v16bf load_frag_bf16(const unsigned short* __restrict__ base,
                                                int row0, int k0, int stride) {
    int lane = threadIdx.x & 31;
    int r    = lane & 15;
    int kb   = (lane & 16) ? 8 : 0;
    const unsigned short* p = base + (size_t)(row0 + r) * stride + k0 + kb;
    FragU f;
    *(uint4v*)(&f.u[0]) = *(const uint4v*)(p);
    *(uint4v*)(&f.u[4]) = *(const uint4v*)(p + 16);
    return f.v;
}

// ---------------- K0a: bulk f32 -> bf16 convert (8 elems/thread) ----------------
__global__ __launch_bounds__(256) void k_cvt_bf16(const float* __restrict__ src,
                                                  unsigned short* __restrict__ dst,
                                                  int n8) {
    int idx = blockIdx.x * blockDim.x + threadIdx.x;
    if (idx >= n8) return;
    const float* p = src + (size_t)idx * 8;
    union { uint4v v; unsigned int u[4]; } o;
#pragma unroll
    for (int i = 0; i < 4; ++i) {
        unsigned int lo = f2bf(p[2 * i]);
        unsigned int hi = f2bf(p[2 * i + 1]);
        o.u[i] = lo | (hi << 16);
    }
    *(uint4v*)(dst + (size_t)idx * 8) = o.v;
}

// ---------------- K0b: transpose + convert weights to bf16 ----------------
// src [K x N] f32 row-major  ->  dst [N x K] bf16 row-major
__global__ __launch_bounds__(256) void k_transpose_cvt(const float* __restrict__ src,
                                                       unsigned short* __restrict__ dst,
                                                       int K, int N) {
    int idx = blockIdx.x * blockDim.x + threadIdx.x;
    if (idx >= K * N) return;
    int k = idx / N, n = idx % N;
    dst[(size_t)n * K + k] = f2bf(src[(size_t)k * N + n]);
}

// ---------------- K1: QKV projection ----------------
// qkv = xb[32768x512] @ w_qkv; 4 M-tiles per wave sharing one B fragment.
// Software-pipelined: next K-step fragments loaded before current wmmas.
__global__ __launch_bounds__(256) void k_qkv(const unsigned short* __restrict__ xb,
                                             const unsigned short* __restrict__ wqkvT,
                                             unsigned short* __restrict__ qb,
                                             unsigned short* __restrict__ kb,
                                             unsigned short* __restrict__ vT) {
    int mb   = blockIdx.x;               // 64-row block (0..511)
    int nb   = blockIdx.y;               // 128-col block (0..11)
    int wave = threadIdx.x >> 5;
    int lane = threadIdx.x & 31;
    int row0 = mb * 64;
    int col0 = nb * 128 + wave * 16;
    v8f acc[4] = {};
    v16bf b_cur = load_frag_bf16(wqkvT, col0, 0, DD);
    v16bf a_cur[4];
#pragma unroll
    for (int t = 0; t < 4; ++t) a_cur[t] = load_frag_bf16(xb, row0 + t * 16, 0, DD);
#pragma unroll
    for (int kk = 0; kk < DD - 32; kk += 32) {
        v16bf b_nxt = load_frag_bf16(wqkvT, col0, kk + 32, DD);
        v16bf a_nxt[4];
#pragma unroll
        for (int t = 0; t < 4; ++t) a_nxt[t] = load_frag_bf16(xb, row0 + t * 16, kk + 32, DD);
#pragma unroll
        for (int t = 0; t < 4; ++t) acc[t] = wmma_bf16(a_cur[t], b_cur, acc[t]);
        b_cur = b_nxt;
#pragma unroll
        for (int t = 0; t < 4; ++t) a_cur[t] = a_nxt[t];
    }
#pragma unroll
    for (int t = 0; t < 4; ++t) acc[t] = wmma_bf16(a_cur[t], b_cur, acc[t]);

    int n = col0 + (lane & 15);
#pragma unroll
    for (int t = 0; t < 4; ++t) {
        int mbase = row0 + t * 16 + ((lane & 16) ? 8 : 0);
#pragma unroll
        for (int r = 0; r < 8; ++r) {
            int m = mbase + r;
            unsigned short h = f2bf(acc[t][r]);
            if (n < DD)            qb[(size_t)m * DD + n] = h;
            else if (n < 2 * DD)   kb[(size_t)m * DD + (n - DD)] = h;
            else {
                int bc = m >> 8, i = m & 255, d = n - 2 * DD;
                vT[((size_t)bc * DD + d) * LL + i] = h;   // v transposed per (b,c)
            }
        }
    }
}

// ---------------- K2: scores + softmax (+ fused attn+pos -> bf16) ----------------
__global__ __launch_bounds__(256) void k_scores(const unsigned short* __restrict__ q,
                                                const unsigned short* __restrict__ k,
                                                const float* __restrict__ pos,
                                                float* __restrict__ attn,
                                                unsigned short* __restrict__ awb) {
    __shared__ float s_tile[16 * LL];
    __shared__ float s_red[16][17];
    int bc = blockIdx.x, rb = blockIdx.y;
    int wave = threadIdx.x >> 5, lane = threadIdx.x & 31;
    int qrow0 = bc * LL + rb * 16;
    int krow0 = bc * LL + wave * 32;     // each wave: 2 column tiles
    v8f acc0 = {}, acc1 = {};
    v16bf a_cur  = load_frag_bf16(q, qrow0, 0, DD);
    v16bf b0_cur = load_frag_bf16(k, krow0,      0, DD);
    v16bf b1_cur = load_frag_bf16(k, krow0 + 16, 0, DD);
#pragma unroll
    for (int kk = 0; kk < DD - 32; kk += 32) {
        v16bf a_nxt  = load_frag_bf16(q, qrow0, kk + 32, DD);
        v16bf b0_nxt = load_frag_bf16(k, krow0,      kk + 32, DD);
        v16bf b1_nxt = load_frag_bf16(k, krow0 + 16, kk + 32, DD);
        acc0 = wmma_bf16(a_cur, b0_cur, acc0);
        acc1 = wmma_bf16(a_cur, b1_cur, acc1);
        a_cur = a_nxt; b0_cur = b0_nxt; b1_cur = b1_nxt;
    }
    acc0 = wmma_bf16(a_cur, b0_cur, acc0);
    acc1 = wmma_bf16(a_cur, b1_cur, acc1);

    int rbase = (lane & 16) ? 8 : 0;
    int c0    = wave * 32 + (lane & 15);
#pragma unroll
    for (int r = 0; r < 8; ++r) {
        s_tile[(rbase + r) * LL + c0]      = acc0[r] * SCALE_F;
        s_tile[(rbase + r) * LL + c0 + 16] = acc1[r] * SCALE_F;
    }
    __syncthreads();
    // row softmax: 16 threads per row
    int row = threadIdx.x >> 4, sub = threadIdx.x & 15;
    float mx = -3.4e38f;
    for (int j = sub; j < LL; j += 16) mx = fmaxf(mx, s_tile[row * LL + j]);
    s_red[row][sub] = mx;
    __syncthreads();
    if (sub == 0) {
        float m2 = s_red[row][0];
        for (int t = 1; t < 16; ++t) m2 = fmaxf(m2, s_red[row][t]);
        s_red[row][16] = m2;
    }
    __syncthreads();
    float rowmax = s_red[row][16];
    float sum = 0.f;
    for (int j = sub; j < LL; j += 16) {
        float e = __expf(s_tile[row * LL + j] - rowmax);
        s_tile[row * LL + j] = e;
        sum += e;
    }
    __syncthreads();
    s_red[row][sub] = sum;
    __syncthreads();
    if (sub == 0) {
        float s2 = 0.f;
        for (int t = 0; t < 16; ++t) s2 += s_red[row][t];
        s_red[row][16] = s2;
    }
    __syncthreads();
    float inv = 1.0f / s_red[row][16];
    int c = bc & 15;
    size_t rowoff = ((size_t)bc * LL + rb * 16 + row) * LL;
    float*          dst  = attn + rowoff;
    unsigned short* dstb = awb  + rowoff;
    const float*    prow = pos + ((size_t)c * LL + rb * 16 + row) * LL;
    for (int j = sub; j < LL; j += 16) {
        float a = s_tile[row * LL + j] * inv;
        dst[j]  = a;
        dstb[j] = f2bf(a + prow[j]);    // pre-fused (attn+pos) for the MLP GEMM
    }
}

// ---------------- K3: per-row MLP -> wvec ----------------
// h = relu(awb @ w1 + b1); wvec = h @ w2 + b2
__global__ __launch_bounds__(256) void k_mlp(const unsigned short* __restrict__ awb,
                                             const unsigned short* __restrict__ w1T,
                                             const float* __restrict__ b1,
                                             const float* __restrict__ w2,
                                             const float* __restrict__ b2,
                                             float* __restrict__ wvec) {
    __shared__ float s_tile[16 * LL];
    __shared__ float s_red[16][17];
    int bc = blockIdx.x, rb = blockIdx.y;
    int wave = threadIdx.x >> 5, lane = threadIdx.x & 31;
    const unsigned short* abase = awb + (size_t)bc * LL * LL;
    int row0 = rb * 16;
    int j0   = wave * 32;
    v8f acc0 = {}, acc1 = {};
    v16bf a_cur  = load_frag_bf16(abase, row0, 0, LL);
    v16bf b0_cur = load_frag_bf16(w1T, j0,      0, LL);
    v16bf b1_cur = load_frag_bf16(w1T, j0 + 16, 0, LL);
#pragma unroll
    for (int kk = 0; kk < LL - 32; kk += 32) {
        v16bf a_nxt  = load_frag_bf16(abase, row0, kk + 32, LL);
        v16bf b0_nxt = load_frag_bf16(w1T, j0,      kk + 32, LL);
        v16bf b1_nxt = load_frag_bf16(w1T, j0 + 16, kk + 32, LL);
        acc0 = wmma_bf16(a_cur, b0_cur, acc0);
        acc1 = wmma_bf16(a_cur, b1_cur, acc1);
        a_cur = a_nxt; b0_cur = b0_nxt; b1_cur = b1_nxt;
    }
    acc0 = wmma_bf16(a_cur, b0_cur, acc0);
    acc1 = wmma_bf16(a_cur, b1_cur, acc1);

    int rbase = (lane & 16) ? 8 : 0;
    int c0    = j0 + (lane & 15);
    float bias0 = b1[c0], bias1 = b1[c0 + 16];
#pragma unroll
    for (int r = 0; r < 8; ++r) {
        s_tile[(rbase + r) * LL + c0]      = fmaxf(acc0[r] + bias0, 0.f);
        s_tile[(rbase + r) * LL + c0 + 16] = fmaxf(acc1[r] + bias1, 0.f);
    }
    __syncthreads();
    int row = threadIdx.x >> 4, sub = threadIdx.x & 15;
    float s = 0.f;
    for (int j = sub; j < LL; j += 16) s += s_tile[row * LL + j] * w2[j];
    s_red[row][sub] = s;
    __syncthreads();
    if (sub == 0) {
        float t = 0.f;
        for (int i2 = 0; i2 < 16; ++i2) t += s_red[row][i2];
        wvec[(size_t)bc * LL + row0 + row] = t + b2[0];
    }
}

// ---------------- K4: Gaussian reweight + second softmax -> attn2 (bf16) ----------------
__global__ __launch_bounds__(256) void k_attn2(const float* __restrict__ attn,
                                               const float* __restrict__ wvec,
                                               unsigned short* __restrict__ attn2) {
    __shared__ float s_tile[16 * LL];
    __shared__ float s_red[16][17];
    int bc = blockIdx.x, rb = blockIdx.y;
    int row = threadIdx.x >> 4, sub = threadIdx.x & 15;
    int i = rb * 16 + row;
    const float* arow = attn + ((size_t)bc * LL + i) * LL;
    float wv  = wvec[(size_t)bc * LL + i];
    float inv2 = 1.0f / (2.0f * wv * wv + 1e-6f);
    float mx = -3.4e38f;
    for (int j = sub; j < LL; j += 16) {
        float d   = (float)(j - i);
        float val = arow[j] * __expf(-(d * d) * inv2);
        s_tile[row * LL + j] = val;
        mx = fmaxf(mx, val);
    }
    s_red[row][sub] = mx;
    __syncthreads();
    if (sub == 0) {
        float m2 = s_red[row][0];
        for (int t = 1; t < 16; ++t) m2 = fmaxf(m2, s_red[row][t]);
        s_red[row][16] = m2;
    }
    __syncthreads();
    float rowmax = s_red[row][16];
    float sum = 0.f;
    for (int j = sub; j < LL; j += 16) {
        float e = __expf(s_tile[row * LL + j] - rowmax);
        s_tile[row * LL + j] = e;
        sum += e;
    }
    __syncthreads();
    s_red[row][sub] = sum;
    __syncthreads();
    if (sub == 0) {
        float s2 = 0.f;
        for (int t = 0; t < 16; ++t) s2 += s_red[row][t];
        s_red[row][16] = s2;
    }
    __syncthreads();
    float inv = 1.0f / s_red[row][16];
    unsigned short* dst = attn2 + ((size_t)bc * LL + i) * LL;
    for (int j = sub; j < LL; j += 16) dst[j] = f2bf(s_tile[row * LL + j] * inv);
}

// ---------------- K5: ctx = attn2 @ v (via vT) ----------------
__global__ __launch_bounds__(256) void k_ctx(const unsigned short* __restrict__ attn2,
                                             const unsigned short* __restrict__ vT,
                                             unsigned short* __restrict__ ctx) {
    int bc = blockIdx.x, rb = blockIdx.y;
    int wave = threadIdx.x >> 5, lane = threadIdx.x & 31;
    const unsigned short* abase = attn2 + (size_t)bc * LL * LL;
    const unsigned short* vbase = vT    + (size_t)bc * DD * LL;
    int row0 = rb * 16;
    int d0   = wave * 64;                 // each wave: 4 column tiles of 16
    v8f acc[4] = {};
    v16bf a_cur = load_frag_bf16(abase, row0, 0, LL);
    v16bf b_cur[4];
#pragma unroll
    for (int t = 0; t < 4; ++t) b_cur[t] = load_frag_bf16(vbase, d0 + t * 16, 0, LL);
#pragma unroll
    for (int kk = 0; kk < LL - 32; kk += 32) {
        v16bf a_nxt = load_frag_bf16(abase, row0, kk + 32, LL);
        v16bf b_nxt[4];
#pragma unroll
        for (int t = 0; t < 4; ++t) b_nxt[t] = load_frag_bf16(vbase, d0 + t * 16, kk + 32, LL);
#pragma unroll
        for (int t = 0; t < 4; ++t) acc[t] = wmma_bf16(a_cur, b_cur[t], acc[t]);
        a_cur = a_nxt;
#pragma unroll
        for (int t = 0; t < 4; ++t) b_cur[t] = b_nxt[t];
    }
#pragma unroll
    for (int t = 0; t < 4; ++t) acc[t] = wmma_bf16(a_cur, b_cur[t], acc[t]);

    int m0 = bc * LL + row0 + ((lane & 16) ? 8 : 0);
#pragma unroll
    for (int t = 0; t < 4; ++t) {
        int n = d0 + t * 16 + (lane & 15);
#pragma unroll
        for (int r = 0; r < 8; ++r)
            ctx[(size_t)(m0 + r) * DD + n] = f2bf(acc[t][r]);
    }
}

// ---------------- K6: out = ctx @ w_out + b_out (f32 result) ----------------
// 4 M-tiles per wave sharing one B fragment; software-pipelined.
__global__ __launch_bounds__(256) void k_out(const unsigned short* __restrict__ ctx,
                                             const unsigned short* __restrict__ woutT,
                                             const float* __restrict__ bout,
                                             float* __restrict__ out) {
    int mb = blockIdx.x;                 // 64-row block (0..511)
    int nb = blockIdx.y;                 // 128-col block (0..3)
    int wave = threadIdx.x >> 5, lane = threadIdx.x & 31;
    int row0 = mb * 64;
    int col0 = nb * 128 + wave * 16;
    v8f acc[4] = {};
    v16bf b_cur = load_frag_bf16(woutT, col0, 0, DD);
    v16bf a_cur[4];
#pragma unroll
    for (int t = 0; t < 4; ++t) a_cur[t] = load_frag_bf16(ctx, row0 + t * 16, 0, DD);
#pragma unroll
    for (int kk = 0; kk < DD - 32; kk += 32) {
        v16bf b_nxt = load_frag_bf16(woutT, col0, kk + 32, DD);
        v16bf a_nxt[4];
#pragma unroll
        for (int t = 0; t < 4; ++t) a_nxt[t] = load_frag_bf16(ctx, row0 + t * 16, kk + 32, DD);
#pragma unroll
        for (int t = 0; t < 4; ++t) acc[t] = wmma_bf16(a_cur[t], b_cur, acc[t]);
        b_cur = b_nxt;
#pragma unroll
        for (int t = 0; t < 4; ++t) a_cur[t] = a_nxt[t];
    }
#pragma unroll
    for (int t = 0; t < 4; ++t) acc[t] = wmma_bf16(a_cur[t], b_cur, acc[t]);

    int n = col0 + (lane & 15);
    float bias = bout[n];
#pragma unroll
    for (int t = 0; t < 4; ++t) {
        int mbase = row0 + t * 16 + ((lane & 16) ? 8 : 0);
#pragma unroll
        for (int r = 0; r < 8; ++r)
            out[(size_t)(mbase + r) * DD + n] = acc[t][r] + bias;
    }
}

// ---------------- Workspace layout (bytes) ----------------
#define OFF_XB     ((size_t)0)
#define OFF_Q      (OFF_XB   + (size_t)MM * DD * 2)          //  33.5 MB
#define OFF_K      (OFF_Q    + (size_t)MM * DD * 2)
#define OFF_VT     (OFF_K    + (size_t)MM * DD * 2)
#define OFF_WQKVT  (OFF_VT   + (size_t)MM * DD * 2)
#define OFF_W1T    (OFF_WQKVT+ (size_t)NQKV * DD * 2)
#define OFF_WOUTT  (OFF_W1T  + (size_t)LL * LL * 2)
#define OFF_ATTN   (OFF_WOUTT+ (size_t)DD * DD * 2)
#define OFF_AWB    (OFF_ATTN + (size_t)BB * CC * LL * LL * 4)
#define OFF_WVEC   (OFF_AWB  + (size_t)BB * CC * LL * LL * 2)
#define OFF_ATTN2  (OFF_WVEC + (size_t)MM * 4)
#define OFF_CTX    (OFF_ATTN2+ (size_t)BB * CC * LL * LL * 2)
// end: OFF_CTX + MM*DD*2  (~237 MB total)

extern "C" void kernel_launch(void* const* d_in, const int* in_sizes, int n_in,
                              void* d_out, int out_size, void* d_ws, size_t ws_size,
                              hipStream_t stream) {
    (void)in_sizes; (void)n_in; (void)out_size; (void)ws_size;
    const float* x     = (const float*)d_in[0];
    const float* w_qkv = (const float*)d_in[1];
    const float* pos   = (const float*)d_in[2];
    const float* w1    = (const float*)d_in[3];
    const float* b1    = (const float*)d_in[4];
    const float* w2    = (const float*)d_in[5];
    const float* b2    = (const float*)d_in[6];
    const float* w_out = (const float*)d_in[7];
    const float* b_out = (const float*)d_in[8];
    float* out = (float*)d_out;

    char* ws = (char*)d_ws;
    unsigned short* xb     = (unsigned short*)(ws + OFF_XB);
    unsigned short* qb     = (unsigned short*)(ws + OFF_Q);
    unsigned short* kb     = (unsigned short*)(ws + OFF_K);
    unsigned short* vT     = (unsigned short*)(ws + OFF_VT);
    unsigned short* wqkvT  = (unsigned short*)(ws + OFF_WQKVT);
    unsigned short* w1T    = (unsigned short*)(ws + OFF_W1T);
    unsigned short* woutT  = (unsigned short*)(ws + OFF_WOUTT);
    float*          attn   = (float*)(ws + OFF_ATTN);
    unsigned short* awb    = (unsigned short*)(ws + OFF_AWB);
    float*          wvec   = (float*)(ws + OFF_WVEC);
    unsigned short* attn2  = (unsigned short*)(ws + OFF_ATTN2);
    unsigned short* ctx    = (unsigned short*)(ws + OFF_CTX);

    // input prep: x -> bf16 (vectorized), weights -> transposed bf16
    k_cvt_bf16<<<(MM * DD / 8 + 255) / 256, 256, 0, stream>>>(x, xb, MM * DD / 8);
    k_transpose_cvt<<<(DD * NQKV + 255) / 256, 256, 0, stream>>>(w_qkv, wqkvT, DD, NQKV);
    k_transpose_cvt<<<(LL * LL   + 255) / 256, 256, 0, stream>>>(w1,    w1T,   LL, LL);
    k_transpose_cvt<<<(DD * DD   + 255) / 256, 256, 0, stream>>>(w_out, woutT, DD, DD);

    // pipeline
    k_qkv   <<<dim3(MM / 64, NQKV / 128), 256, 0, stream>>>(xb, wqkvT, qb, kb, vT);
    k_scores<<<dim3(BB * CC, LL / 16),    256, 0, stream>>>(qb, kb, pos, attn, awb);
    k_mlp   <<<dim3(BB * CC, LL / 16),    256, 0, stream>>>(awb, w1T, b1, w2, b2, wvec);
    k_attn2 <<<dim3(BB * CC, LL / 16),    256, 0, stream>>>(attn, wvec, attn2);
    k_ctx   <<<dim3(BB * CC, LL / 16),    256, 0, stream>>>(attn2, vT, ctx);
    k_out   <<<dim3(MM / 64, DD / 128),   256, 0, stream>>>(ctx, woutT, b_out, out);
}
